// Srf_layer_2_8521215115645
// MI455X (gfx1250) — matrix operand order repaired and checked by reading, not compiled, not run on hardware
//
#include <hip/hip_runtime.h>
#include <cmath>

// ---------------------------------------------------------------------------
// SRF grouped conv for MI455X (gfx1250, wave32, WMMA + TDM).
//
//  data   [16,20,384,384] f32   (d_in[0])
//  alphas [5,6,4,8]       f32   (d_in[1])
//  scales [5]             f32   (d_in[2])
//  out    [16,40,384,384] f32   (d_out)
//
// Roofline: 18.9 GFLOP vs 566 MB HBM traffic -> memory bound (~24us floor at
// 23.3 TB/s) IF math runs on the matrix pipe. HBM traffic stays fp32
// (reference volume); contraction runs on v_wmma_f32_16x16x32_f16
// (f16 operands staged in LDS, f32 accumulate).
//
// GEMM mapping per group g (4 in-ch, 8 out-ch, 5x5 taps):
//   M = 16 out-channels (8 real, rows 8..15 zero weights)
//   N = 16 output pixels (one row segment)
//   K = 128 = inCh(4) * 32  (tap index t = ky*5+kx, t in [0,25), pad to 32)
// Each wave: 4-step WMMA chain (one 16x16x32 step per input channel).
// Block: 256 thr = 8 waves -> 8x32 output tile for one (batch, group).
//
// CDNA5-specific paths exercised:
//   * v_wmma_f32_16x16x32_f16   (matrix pipe)
//   * tensor_load_to_lds + s_wait_tensorcnt  (TDM DMA of the f16 weight tile,
//     overlapped with the fp32->f16 input-halo staging)
//   * global_prefetch_b8        (next-tile halo prefetch)
// ---------------------------------------------------------------------------

typedef __attribute__((ext_vector_type(16))) _Float16 v16h;
typedef __attribute__((ext_vector_type(8)))  float    v8f;
typedef unsigned int u32x4 __attribute__((ext_vector_type(4)));
typedef int          i32x8 __attribute__((ext_vector_type(8)));
typedef int          i32x4 __attribute__((ext_vector_type(4)));

#define NSCALES 5
#define INCH    20
#define OUTCH   40
#define GRP_IC  4
#define GRP_OC  8
#define IMH     384
#define IMW     384
#define NBATCH  16
#define KPAD    128          // 4 channels * 32 padded taps
#define TY      8            // output tile rows per block
#define TX      32           // output tile cols per block
#define IN_ROWS (TY + 4)     // 12 (halo for 5x5)
#define IN_COLS (TX + 4)     // 36
#define IN_PITCH 40          // padded LDS pitch

// ---------------------------------------------------------------------------
// Kernel 1: build padded f16 weight buffer wbuf[g][m=16][k=128] in d_ws.
//   m in [0,8): real out-channel, else zero.  k = c*32 + t, t<25 real tap.
//   Gaussian-Hermite basis, order 2, learn_sigma path (tanh).
// ---------------------------------------------------------------------------
__global__ __launch_bounds__(256) void srf_make_weights(
    const float* __restrict__ alphas,   // [5][6][4][8]
    const float* __restrict__ scales,   // [5]
    _Float16* __restrict__ wbuf)        // [5][16][128]
{
  int idx = blockIdx.x * 256 + threadIdx.x;          // 5*16*128 = 10240 threads
  if (idx >= NSCALES * 16 * KPAD) return;
  int g = idx / (16 * KPAD);
  int m = (idx / KPAD) & 15;
  int k = idx & (KPAD - 1);
  int c = k >> 5;            // input channel within group
  int t = k & 31;            // padded tap index

  float val = 0.0f;
  if (m < GRP_OC && t < 25) {
    int fs  = (g < 3) ? 1 : 2;                 // FS = ceil(2*sigma0) = [1,1,1,2,2]
    int ksz = 2 * fs + 1;                      // 3 or 5
    int p   = (5 - ksz) >> 1;                  // zero-pad border to 5x5
    int ky  = t / 5 - p;
    int kx  = t % 5 - p;
    if (ky >= 0 && ky < ksz && kx >= 0 && kx < ksz) {
      float mn    = 0.2f * (float)g;
      float mx    = mn + 0.2f;
      float sigma = 0.5f * (mx - mn) * tanhf(scales[g]) + 0.5f * (mn + mx);
      float s2    = sigma * sigma;
      // unit-sum Gaussian envelope normalizer over the small kernel
      float norm = 0.0f;
      for (int xx = -fs; xx <= fs; ++xx)
        norm += expf(-(float)(xx * xx) / (2.0f * s2));
      float xk = (float)(ky - fs);
      float xl = (float)(kx - fs);
      float gk = expf(-xk * xk / (2.0f * s2)) / norm;
      float gl = expf(-xl * xl / (2.0f * s2)) / norm;
      // n-th Gaussian derivatives: d0=g, d1=-(x/s^2)g, d2=(x^2/s^4 - 1/s^2)g
      float dk[3] = { gk, -(xk / s2) * gk, (xk * xk / (s2 * s2) - 1.0f / s2) * gk };
      float dl[3] = { gl, -(xl / s2) * gl, (xl * xl / (s2 * s2) - 1.0f / s2) * gl };
      // basis order: (i,j) with i+j<=2 in python loop order
      const int bi[6] = {0, 0, 0, 1, 1, 2};
      const int bj[6] = {0, 1, 2, 0, 1, 0};
      #pragma unroll
      for (int n = 0; n < 6; ++n) {
        float a = alphas[((g * 6 + n) * GRP_IC + c) * GRP_OC + m];
        val += dk[bi[n]] * dl[bj[n]] * a;
      }
    }
  }
  wbuf[idx] = (_Float16)val;
}

// ---------------------------------------------------------------------------
// Kernel 2: WMMA grouped conv, weights DMA'd by the Tensor Data Mover.
// grid = (48*12 tiles, 5 groups, 16 batch), block = 256 (8 waves).
// ---------------------------------------------------------------------------
__global__ __launch_bounds__(256) void srf_conv_wmma(
    const float* __restrict__ in,       // [16][20][384][384]
    const _Float16* __restrict__ wbuf,  // [5][16][128]
    float* __restrict__ out)            // [16][40][384][384]
{
  __shared__ _Float16 lds_in[GRP_IC][IN_ROWS][IN_PITCH]; // 3840 B
  __shared__ _Float16 lds_w[16][KPAD];                   // 4096 B

  const int tid = threadIdx.x;
  const int g   = blockIdx.y;
  const int b   = blockIdx.z;
  const int tiles_x = IMW / TX;                 // 12
  const int ty  = blockIdx.x / tiles_x;
  const int tx  = blockIdx.x % tiles_x;
  const int y0  = ty * TY;
  const int x0  = tx * TX;

  // ---- TDM: DMA the 16x128 f16 weight tile (4 KB, contiguous) into LDS ----
  // Descriptor per ISA 8.3/8.4: 1D tensor of 2048 2-byte elements, tile ==
  // tensor. Groups 2/3 zero (<=2D tensor). Issued by one wave (EXEC-gated
  // branch); TENSORcnt is per-wave, so wave 0 waits then all sync at barrier.
  if (tid == 0) {
    unsigned lds_off = (unsigned)(size_t)(&lds_w[0][0]);
    unsigned long long ga =
        (unsigned long long)(const void*)(wbuf + (size_t)g * 16 * KPAD);
    u32x4 g0 = { 1u,                                  // count=1 (valid D#)
                 lds_off,                             // lds_addr (bytes)
                 (unsigned)ga,                        // global_addr[31:0]
                 (unsigned)((ga >> 32) & 0x1FFFFFFu)  // global_addr[56:32]
                     | 0x80000000u };                 // type=2 ("image")
    i32x8 g1 = { 0x00010000,            // workgroup_mask=0, data_size=1 (2B)
                 (int)(2048u << 16),    // tensor_dim0 = 2048 (lo16 in [31:16])
                 0x00010000,            // tensor_dim0 hi=0, tensor_dim1 = 1
                 (int)(2048u << 16),    // tensor_dim1 hi=0, tile_dim0 = 2048
                 0,                     // tile_dim1 = 0 (unused), tile_dim2 = 0
                 2048,                  // tensor_dim0_stride[31:0]
                 0, 0 };
    i32x4 gz4 = { 0, 0, 0, 0 };
    i32x8 gz8 = { 0, 0, 0, 0, 0, 0, 0, 0 };
    __builtin_amdgcn_tensor_load_to_lds(g0, g1, gz4, gz4, gz8, /*cpol=*/0);
  }

  // ---- stage fp32 input halo tile -> f16 LDS (overlaps the TDM copy) ----
  const long ch_base = ((long)b * INCH + g * GRP_IC) * (long)(IMH * IMW);
  for (int i = tid; i < GRP_IC * IN_ROWS * IN_COLS; i += 256) {
    int c   = i / (IN_ROWS * IN_COLS);
    int r   = (i / IN_COLS) % IN_ROWS;
    int col = i % IN_COLS;
    int gy = y0 - 2 + r;
    int gx = x0 - 2 + col;
    float v = 0.0f;
    if (gy >= 0 && gy < IMH && gx >= 0 && gx < IMW)
      v = in[ch_base + (long)c * (IMH * IMW) + (long)gy * IMW + gx];
    lds_in[c][r][col] = (_Float16)v;
  }

  // ---- speculative prefetch of the next x-tile's halo rows into caches ----
  if (tx + 1 < tiles_x) {
    for (int i = tid; i < GRP_IC * IN_ROWS; i += 256) {
      int c = i / IN_ROWS;
      int r = i % IN_ROWS;
      int gy = y0 - 2 + r;
      if (gy >= 0 && gy < IMH)
        __builtin_prefetch(
            &in[ch_base + (long)c * (IMH * IMW) + (long)gy * IMW + (x0 + TX)],
            0, 1);
    }
  }

  // wave 0 waits for its TDM transfer; other waves have TENSORcnt==0 (no-op)
  __builtin_amdgcn_s_wait_tensorcnt(0);
  __syncthreads();

  // wave tiling: wave w -> output row (w>>1), column segment ((w&1)*16)
  const int wave   = tid >> 5;
  const int lane   = tid & 31;
  const int row    = wave >> 1;          // 0..3 within tile
  const int colseg = (wave & 1) * 16;    // 0 or 16
  const int ml     = lane & 15;          // M row (A) / N col (B) per ISA layout
  const int hi     = lane >> 4;          // K-half select (contiguous k order)

  union Frag { v16h v; _Float16 h[16]; };

  v8f acc = {};
  #pragma unroll
  for (int c = 0; c < GRP_IC; ++c) {
    // ---- A fragment: weights, 16x32 f16 (contiguous k order, same as B) ----
    // lane<16: K in {0..15}; lane>=16: K in {16..31}
    Frag a;
    #pragma unroll
    for (int v = 0; v < 8; ++v) {
      int kb = 16 * hi + 2 * v;
      a.h[2 * v]     = lds_w[ml][c * 32 + kb];
      a.h[2 * v + 1] = lds_w[ml][c * 32 + kb + 1];
    }
    // ---- B fragment: im2col pixels, 32x16 f16 ----
    // lane gives N (pixel), lane-half gives K range 0-15 / 16-31
    Frag bf;
    const int xloc = colseg + ml;        // local output column of this lane
    #pragma unroll
    for (int v = 0; v < 8; ++v) {
      #pragma unroll
      for (int u = 0; u < 2; ++u) {
        int t = hi * 16 + 2 * v + u;     // padded tap index 0..31
        _Float16 val = (_Float16)0.0f;
        if (t < 25)
          val = lds_in[c][row + t / 5][xloc + t % 5];
        bf.h[2 * v + u] = val;
      }
    }
    // D = A x B + C, f32 accumulate
    acc = __builtin_amdgcn_wmma_f32_16x16x32_f16(
        /*neg_a=*/false, a.v, /*neg_b=*/false, bf.v,
        /*c_mod=*/(short)0, acc, /*reuse_a=*/false, /*reuse_b=*/false);
  }

  // ---- store: C/D layout -> lanes 0-15 hold M rows 0..7 (real channels),
  //      lanes 16-31 hold rows 8..15 (zero-padded) -> nothing to store.
  if (lane < 16) {
    int y = y0 + row;
    int x = x0 + colseg + lane;
    long base = (((long)b * OUTCH + g * GRP_OC) * IMH + y) * (long)IMW + x;
    #pragma unroll
    for (int r = 0; r < 8; ++r)
      out[base + (long)r * (IMH * IMW)] = acc[r];
  }
}

// ---------------------------------------------------------------------------
extern "C" void kernel_launch(void* const* d_in, const int* in_sizes, int n_in,
                              void* d_out, int out_size, void* d_ws, size_t ws_size,
                              hipStream_t stream) {
  const float* data   = (const float*)d_in[0];
  const float* alphas = (const float*)d_in[1];
  const float* scales = (const float*)d_in[2];
  float* out = (float*)d_out;
  _Float16* wbuf = (_Float16*)d_ws;     // 5*16*128 f16 = 20 KB

  // 1) build padded f16 filter bank
  srf_make_weights<<<(NSCALES * 16 * KPAD) / 256, 256, 0, stream>>>(
      alphas, scales, wbuf);

  // 2) grouped conv via WMMA + TDM
  dim3 grid((IMH / TY) * (IMW / TX), NSCALES, NBATCH);  // (576, 5, 16)
  srf_conv_wmma<<<grid, 256, 0, stream>>>(data, wbuf, out);
}
